// DiTBlock_3126736192253
// MI455X (gfx1250) — matrix operand order repaired and checked by reading, not compiled, not run on hardware
//
#include <hip/hip_runtime.h>
#include <hip/hip_bf16.h>
#include <math.h>

typedef __bf16 bf16;
typedef __attribute__((ext_vector_type(16))) __bf16 v16bf;
typedef __attribute__((ext_vector_type(8)))  float  v8f;
typedef __attribute__((ext_vector_type(4)))  int    v4i_t;

#define WMMA_BF16(a,b,c) __builtin_amdgcn_wmma_f32_16x16x32_bf16(false,(a),false,(b),(short)0,(c),false,false)

// ---- async global->LDS (CDNA5, ASYNCcnt) with sync fallback ----
#if defined(__has_builtin)
#  if __has_builtin(__builtin_amdgcn_global_load_async_to_lds_b128)
#    define HAVE_ASYNC 1
#  endif
#endif
#ifndef HAVE_ASYNC
#  define HAVE_ASYNC 0
#endif

#if HAVE_ASYNC
#define AS1 __attribute__((address_space(1)))
#define AS3 __attribute__((address_space(3)))
static __device__ inline void cp16(bf16* lds_dst, const bf16* g_src){
  __builtin_amdgcn_global_load_async_to_lds_b128(
      (AS1 v4i_t*)(v4i_t*)g_src, (AS3 v4i_t*)(v4i_t*)lds_dst, 0, 0);
}
static __device__ inline void wait_async(){
#if __has_builtin(__builtin_amdgcn_s_wait_asynccnt)
  __builtin_amdgcn_s_wait_asynccnt(0);
#else
  asm volatile("s_wait_asynccnt 0x0" ::: "memory");
#endif
}
#else
static __device__ inline void cp16(bf16* lds_dst, const bf16* g_src){
  *(int4*)lds_dst = *(const int4*)g_src;
}
static __device__ inline void wait_async(){}
#endif

// ---- fragment loaders (layouts per CDNA5 ISA §7.12.2) ----
static __device__ inline v16bf frag_a(const bf16* base, int stride, int lane){
  int m = lane & 15, hi = lane >> 4;
  union { v16bf v; unsigned u[8]; } r;
  const bf16* row = base + m * stride;
#pragma unroll
  for (int i = 0; i < 8; ++i){
    int k = (i < 4 ? 2*i : 16 + 2*(i-4)) + 8*hi;
    r.u[i] = *(const unsigned*)(row + k);
  }
  return r.v;
}
static __device__ inline v16bf frag_b(const bf16* base, int stride, int lane){
  int n = lane & 15, hi = lane >> 4;
  union { v16bf v; unsigned u[8]; } r;
  const bf16* row = base + n * stride;
#pragma unroll
  for (int i = 0; i < 8; ++i){
    int k = (i < 4 ? 2*i : 16 + 2*(i-4)) + 8*hi;
    r.u[i] = *(const unsigned*)(row + k);
  }
  return r.v;
}

// ---- weight transpose + bf16 convert: WT[n*K+k] = W[k*Nn+n] ----
__global__ void twb_kernel(const float* W, bf16* WT, int Kd, int Nn){
  int idx = blockIdx.x * 256 + threadIdx.x;
  if (idx >= Kd * Nn) return;
  int n = idx / Kd, k = idx - n * Kd;
  WT[idx] = (bf16)W[(size_t)k * Nn + n];
}

// ---- adaLN: ada[b][n] = silu(cond[b]) . ada_w[:,n] + ada_b[n] ----
__global__ void ada_kernel(const float* cond, const float* w, const float* bias, float* ada){
  int idx = blockIdx.x * 256 + threadIdx.x;   // 0..2*3072-1
  int b = idx / 3072, n = idx - b * 3072;
  const float* cb = cond + b * 512;
  float acc = bias[n];
  for (int k = 0; k < 512; ++k){
    float c = cb[k];
    c = c / (1.f + __expf(-c));
    acc += c * w[(size_t)k * 3072 + n];
  }
  ada[idx] = acc;
}

// ---- modulated LayerNorm -> bf16: out = ln(x)*(1+sc)+sh ----
__global__ __launch_bounds__(256) void modln_kernel(const float* X, const float* ada,
                                                    int sh_off, int sc_off, bf16* Out){
  int row = blockIdx.x;            // 0..8191
  int b = row >> 12;               // N = 4096
  const float* xr = X + (size_t)row * 512;
  int t = threadIdx.x;
  float v0 = xr[t], v1 = xr[t + 256];
  float s = v0 + v1, s2 = v0*v0 + v1*v1;
#pragma unroll
  for (int off = 16; off; off >>= 1){
    s  += __shfl_xor(s,  off, 32);
    s2 += __shfl_xor(s2, off, 32);
  }
  __shared__ float red[16];
  int lane = t & 31, w = t >> 5;
  if (lane == 0){ red[w] = s; red[8 + w] = s2; }
  __syncthreads();
  float S = 0.f, S2 = 0.f;
#pragma unroll
  for (int i = 0; i < 8; ++i){ S += red[i]; S2 += red[8 + i]; }
  float mean = S * (1.0f/512.0f);
  float var  = S2 * (1.0f/512.0f) - mean*mean;
  float inv  = rsqrtf(var + 1e-6f);
  const float* adab = ada + b * 3072;
  bf16* outr = Out + (size_t)row * 512;
  float sc0 = 1.f + adab[sc_off + t],       sh0 = adab[sh_off + t];
  float sc1 = 1.f + adab[sc_off + t + 256], sh1 = adab[sh_off + t + 256];
  outr[t]       = (bf16)((v0 - mean) * inv * sc0 + sh0);
  outr[t + 256] = (bf16)((v1 - mean) * inv * sc1 + sh1);
}

// ---- generic WMMA GEMM, double-buffered LDS + async staging ----
struct GemmArgs {
  const bf16* A; const bf16* WT; const float* bias;
  const float* ada;     // gate vector source (PROJ/MLP2)
  const float* resid;   // residual input (PROJ: x, MLP2: x1)
  float* out_f32;       // PROJ: x1, MLP2: final out
  bf16*  out_bf16;      // GELU output
  bf16 *q, *k, *v;      // QKV outputs [B,H,N,D]
  int M, Nmat, K, gate_off;
};

template<int MODE>   // 0=QKV scatter, 1=PROJ(+gate+resid), 2=GELU, 3=MLP2(+gate+resid)
__global__ __launch_bounds__(256) void gemm_kernel(GemmArgs ga){
  __shared__ bf16 As[2][128 * 40];   // stride 40: 80B rows, 16B aligned, conflict-free
  __shared__ bf16 Bs[2][64 * 40];
  const int t = threadIdx.x, lane = t & 31, w = t >> 5;
  const int m_base = blockIdx.x * 128, n_base = blockIdx.y * 64;
  // per-thread staging coords
  const int ar = t >> 1, ah = t & 1;       // A: 128 rows x 2 halves, 16 bf16 each
  const int bn = t >> 2, bq = t & 3;       // B: 64 rows x 4 quarters, 8 bf16 each
  const bf16* aG = ga.A  + (size_t)(m_base + ar) * ga.K + ah * 16;
  const bf16* bG = ga.WT + (size_t)(n_base + bn) * ga.K + bq * 8;
  bf16* aL = &As[0][ar * 40 + ah * 16];
  bf16* bL = &Bs[0][bn * 40 + bq * 8];
  const int ABUF = 128 * 40, BBUF = 64 * 40;

  auto stage = [&](int buf, int k0){
    cp16(aL + buf * ABUF,     aG + k0);
    cp16(aL + buf * ABUF + 8, aG + k0 + 8);
    cp16(bL + buf * BBUF,     bG + k0);
  };

  v8f acc[4] = {};
  const int nch = ga.K >> 5;
  stage(0, 0);
  for (int kc = 0; kc < nch; ++kc){
    int buf = kc & 1;
    wait_async();
    __syncthreads();
    if (kc + 1 < nch) stage(buf ^ 1, (kc + 1) * 32);   // overlaps compute below
    v16bf a = frag_a(&As[buf][w * 16 * 40], 40, lane);
#pragma unroll
    for (int j = 0; j < 4; ++j)
      acc[j] = WMMA_BF16(a, frag_b(&Bs[buf][j * 16 * 40], 40, lane), acc[j]);
  }

  const int hi = lane >> 4, l15 = lane & 15;
#pragma unroll
  for (int j = 0; j < 4; ++j){
    int ng = n_base + j * 16 + l15;
    float bs = ga.bias[ng];
#pragma unroll
    for (int r = 0; r < 8; ++r){
      int mg = m_base + w * 16 + r + 8 * hi;   // C-layout: M = r + 8*(lane>=16)
      float val = acc[j][r] + bs;
      if (MODE == 0){                 // scatter to q/k/v [B,H,N,64] bf16
        int part = ng >> 9, c = ng & 511, hh = c >> 6, d = c & 63;
        int bb = mg >> 12, rowN = mg & 4095;
        bf16* dst = (part == 0) ? ga.q : (part == 1) ? ga.k : ga.v;
        dst[((size_t)((bb * 8 + hh) * 4096 + rowN)) * 64 + d] = (bf16)val;
      } else if (MODE == 2){          // exact GELU
        val = 0.5f * val * (1.0f + erff(val * 0.70710678118f));
        ga.out_bf16[(size_t)mg * 512 + ng] = (bf16)val;
      } else {                        // PROJ / MLP2: resid + gate*val
        int bb = mg >> 12;
        float g = ga.ada[bb * 3072 + ga.gate_off + ng];
        ga.out_f32[(size_t)mg * 512 + ng] = ga.resid[(size_t)mg * 512 + ng] + g * val;
      }
    }
  }
}

// ---- flash attention: (bh, 128-row tile); 8 waves x 16 rows; K/V double-buffered ----
__global__ __launch_bounds__(256) void attn_kernel(const bf16* Q, const bf16* K, const bf16* V,
                                                   const int* mask, bf16* Out, float scale){
  __shared__ bf16 Qs[128 * 72];      // after q-frag loads, wave w's 16 rows are reused for P
  __shared__ bf16 Ks[2][64 * 72];
  __shared__ bf16 Vt[2][64 * 72];    // transposed: Vt[d][key]
  const int t = threadIdx.x, lane = t & 31, w = t >> 5;
  const int hi = lane >> 4, l15 = lane & 15;
  const int bh = blockIdx.y, b = bh >> 3, h = bh & 7;
  const int row0 = blockIdx.x * 128;
  const bf16* Qbh = Q + (size_t)bh * 4096 * 64;
  const bf16* Kbh = K + (size_t)bh * 4096 * 64;
  const bf16* Vbh = V + (size_t)bh * 4096 * 64;

  // per-thread K/V staging coords: 64 rows x 4 quarters of 16 bf16
  const int kr = t >> 2, kq = t & 3;
  const bf16* kG = Kbh + (size_t)kr * 64 + kq * 16;
  const bf16* vG = Vbh + (size_t)kr * 64 + kq * 16;
  bf16* kL = &Ks[0][kr * 72 + kq * 16];
  const int KBUF = 64 * 72;

  auto stageK = [&](int buf, int kc){
    cp16(kL + buf * KBUF,     kG + (size_t)kc * 64 * 64);
    cp16(kL + buf * KBUF + 8, kG + (size_t)kc * 64 * 64 + 8);
  };
  int4 va, vb;
  auto loadV = [&](int kc){
    va = *(const int4*)(vG + (size_t)kc * 64 * 64);
    vb = *(const int4*)(vG + (size_t)kc * 64 * 64 + 8);
  };
  auto storeV = [&](int buf){
    bf16 tmp[16];
    *(int4*)tmp = va; *(int4*)(tmp + 8) = vb;
#pragma unroll
    for (int i = 0; i < 16; ++i) Vt[buf][(kq * 16 + i) * 72 + kr] = tmp[i];
  };

  { // stage Q tile 128x64 (sync)
    int r = t >> 1, half = t & 1;
    const int4* src = (const int4*)(Qbh + (size_t)(row0 + r) * 64 + half * 32);
    int4* dst = (int4*)(Qs + r * 72 + half * 32);
    dst[0] = src[0]; dst[1] = src[1];
  }
  stageK(0, 0);
  loadV(0);
  storeV(0);
  wait_async();
  __syncthreads();

  v16bf qa0 = frag_a(Qs + w * 16 * 72,      72, lane);
  v16bf qa1 = frag_a(Qs + w * 16 * 72 + 32, 72, lane);
  bf16* Pw = Qs + w * 16 * 72;   // wave-private P region (aliases this wave's Q rows)

  v8f o[4] = {};
  float mrow[8], lrow[8];
#pragma unroll
  for (int r = 0; r < 8; ++r){ mrow[r] = -1e30f; lrow[r] = 0.f; }

  for (int kc = 0; kc < 64; ++kc){
    int buf = kc & 1;
    if (kc + 1 < 64){ stageK(buf ^ 1, kc + 1); loadV(kc + 1); }  // overlaps compute
    float biasv[4];
#pragma unroll
    for (int j = 0; j < 4; ++j){
      int key = kc * 64 + j * 16 + l15;
      biasv[j] = (mask[b * 4096 + key] != 1) ? -10000.f : 0.f;
    }
    // S = Q K^T (2 k-chunks of 32 over D=64)
    v8f s[4];
#pragma unroll
    for (int j = 0; j < 4; ++j){
      v8f z = {};
      z = WMMA_BF16(qa0, frag_b(&Ks[buf][j * 16 * 72],      72, lane), z);
      z = WMMA_BF16(qa1, frag_b(&Ks[buf][j * 16 * 72 + 32], 72, lane), z);
      s[j] = z;
    }
#pragma unroll
    for (int j = 0; j < 4; ++j)
#pragma unroll
      for (int r = 0; r < 8; ++r)
        s[j][r] = s[j][r] * scale + biasv[j];
    // online softmax (16-lane reductions match C-layout M = r + 8*(lane>=16))
    float mnew[8], alpha[8];
#pragma unroll
    for (int r = 0; r < 8; ++r){
      float mx = fmaxf(fmaxf(s[0][r], s[1][r]), fmaxf(s[2][r], s[3][r]));
#pragma unroll
      for (int off = 8; off; off >>= 1) mx = fmaxf(mx, __shfl_xor(mx, off, 32));
      mnew[r] = fmaxf(mrow[r], mx);
      alpha[r] = __expf(mrow[r] - mnew[r]);
      mrow[r] = mnew[r];
    }
#pragma unroll
    for (int j = 0; j < 4; ++j)
#pragma unroll
      for (int r = 0; r < 8; ++r)
        s[j][r] = __expf(s[j][r] - mnew[r]);
#pragma unroll
    for (int r = 0; r < 8; ++r){
      float sm = s[0][r] + s[1][r] + s[2][r] + s[3][r];
#pragma unroll
      for (int off = 8; off; off >>= 1) sm += __shfl_xor(sm, off, 32);
      lrow[r] = lrow[r] * alpha[r] + sm;
    }
#pragma unroll
    for (int jd = 0; jd < 4; ++jd)
#pragma unroll
      for (int r = 0; r < 8; ++r)
        o[jd][r] *= alpha[r];
    // P: C-layout -> A-layout via wave-private LDS region (no barrier needed)
#pragma unroll
    for (int j = 0; j < 4; ++j)
#pragma unroll
      for (int r = 0; r < 8; ++r)
        Pw[(r + 8 * hi) * 72 + j * 16 + l15] = (bf16)s[j][r];
    // O += P @ V
#pragma unroll
    for (int c = 0; c < 2; ++c){
      v16bf pa = frag_a(Pw + c * 32, 72, lane);
#pragma unroll
      for (int jd = 0; jd < 4; ++jd)
        o[jd] = WMMA_BF16(pa, frag_b(&Vt[buf][jd * 16 * 72 + c * 32], 72, lane), o[jd]);
    }
    if (kc + 1 < 64) storeV(buf ^ 1);   // V regs ready; other waves done with buf^1
    wait_async();                        // next K chunk landed
    __syncthreads();
  }
  // normalize + store into [B*N, C] bf16 (head h occupies cols h*64..)
#pragma unroll
  for (int jd = 0; jd < 4; ++jd)
#pragma unroll
    for (int r = 0; r < 8; ++r){
      int rowN = row0 + w * 16 + r + 8 * hi;
      int col  = h * 64 + jd * 16 + l15;
      float val = o[jd][r] / lrow[r];
      Out[((size_t)(b * 4096 + rowN)) * 512 + col] = (bf16)val;
    }
}

extern "C" void kernel_launch(void* const* d_in, const int* in_sizes, int n_in,
                              void* d_out, int out_size, void* d_ws, size_t ws_size,
                              hipStream_t stream){
  const float* x      = (const float*)d_in[0];
  const float* cond   = (const float*)d_in[1];
  const int*   mask   = (const int*)  d_in[2];
  const float* qkv_w  = (const float*)d_in[3];
  const float* qkv_b  = (const float*)d_in[4];
  const float* proj_w = (const float*)d_in[5];
  const float* proj_b = (const float*)d_in[6];
  const float* ada_w  = (const float*)d_in[7];
  const float* ada_b  = (const float*)d_in[8];
  const float* mlp_w1 = (const float*)d_in[9];
  const float* mlp_b1 = (const float*)d_in[10];
  const float* mlp_w2 = (const float*)d_in[11];
  const float* mlp_b2 = (const float*)d_in[12];
  float* out = (float*)d_out;

  char* p = (char*)d_ws;
  auto alloc = [&](size_t bytes)->char*{ char* r = p; p += (bytes + 255) & ~(size_t)255; return r; };
  bf16*  qkv_wT = (bf16*) alloc((size_t)512 * 1536 * 2);
  bf16*  proj_wT= (bf16*) alloc((size_t)512 * 512 * 2);
  bf16*  w1T    = (bf16*) alloc((size_t)512 * 512 * 2);
  bf16*  w2T    = (bf16*) alloc((size_t)512 * 512 * 2);
  float* ada    = (float*)alloc((size_t)2 * 3072 * 4);
  bf16*  actA   = (bf16*) alloc((size_t)8192 * 512 * 2);   // h / h_mlp
  bf16*  actB   = (bf16*) alloc((size_t)8192 * 512 * 2);   // attn_out / gelu_out
  bf16*  qb     = (bf16*) alloc((size_t)8192 * 512 * 2);
  bf16*  kb     = (bf16*) alloc((size_t)8192 * 512 * 2);
  bf16*  vb     = (bf16*) alloc((size_t)8192 * 512 * 2);
  float* x1     = (float*)alloc((size_t)8192 * 512 * 4);
  (void)ws_size; (void)n_in; (void)in_sizes; (void)out_size;

  twb_kernel<<<(512*1536 + 255)/256, 256, 0, stream>>>(qkv_w, qkv_wT, 512, 1536);
  twb_kernel<<<(512*512  + 255)/256, 256, 0, stream>>>(proj_w, proj_wT, 512, 512);
  twb_kernel<<<(512*512  + 255)/256, 256, 0, stream>>>(mlp_w1, w1T, 512, 512);
  twb_kernel<<<(512*512  + 255)/256, 256, 0, stream>>>(mlp_w2, w2T, 512, 512);
  ada_kernel<<<24, 256, 0, stream>>>(cond, ada_w, ada_b, ada);

  // attn branch: h = ln(x)*(1+sc_msa)+sh_msa   (chunk order: sh,sc,g,sh,sc,g)
  modln_kernel<<<8192, 256, 0, stream>>>(x, ada, /*sh*/0, /*sc*/512, actA);

  GemmArgs gq{}; gq.A = actA; gq.WT = qkv_wT; gq.bias = qkv_b;
  gq.q = qb; gq.k = kb; gq.v = vb; gq.M = 8192; gq.Nmat = 1536; gq.K = 512;
  gemm_kernel<0><<<dim3(64, 24), 256, 0, stream>>>(gq);

  attn_kernel<<<dim3(32, 16), 256, 0, stream>>>(qb, kb, vb, mask, actB, 0.125f);

  GemmArgs gp{}; gp.A = actB; gp.WT = proj_wT; gp.bias = proj_b;
  gp.ada = ada; gp.resid = x; gp.out_f32 = x1; gp.gate_off = 1024;  // g_msa = 2C
  gp.M = 8192; gp.Nmat = 512; gp.K = 512;
  gemm_kernel<1><<<dim3(64, 8), 256, 0, stream>>>(gp);

  // MLP branch
  modln_kernel<<<8192, 256, 0, stream>>>(x1, ada, /*sh*/1536, /*sc*/2048, actA);

  GemmArgs g1{}; g1.A = actA; g1.WT = w1T; g1.bias = mlp_b1;
  g1.out_bf16 = actB; g1.M = 8192; g1.Nmat = 512; g1.K = 512;
  gemm_kernel<2><<<dim3(64, 8), 256, 0, stream>>>(g1);

  GemmArgs g2{}; g2.A = actB; g2.WT = w2T; g2.bias = mlp_b2;
  g2.ada = ada; g2.resid = x1; g2.out_f32 = out; g2.gate_off = 2560; // g_mlp = 5C
  g2.M = 8192; g2.Nmat = 512; g2.K = 512;
  gemm_kernel<3><<<dim3(64, 8), 256, 0, stream>>>(g2);
}
